// NxDMoE_45956150067870
// MI455X (gfx1250) — compile-verified, matrix operand order checked by reading, not run on hardware
//
#include <hip/hip_runtime.h>
#include <hip/hip_bf16.h>

// Problem constants (from reference)
#define E_ 16
#define K_ 4
#define H_ 3072
#define I_ 3072
#define C_ 256
#define T_ 512
#define ALPHA_ 1.702f
#define LIMIT_ 7.0f

#define USE_ASYNC_COPY 1

typedef __bf16 v16bf __attribute__((ext_vector_type(16)));
typedef float  v8f   __attribute__((ext_vector_type(8)));

__device__ __forceinline__ unsigned short bf16_rne(float f) {
  union { float f; unsigned u; } c; c.f = f;
  unsigned u = c.u;
  return (unsigned short)((u + 0x7FFFu + ((u >> 16) & 1u)) >> 16);
}

__device__ __forceinline__ unsigned pack_bf16(float lo, float hi) {
  return (unsigned)bf16_rne(lo) | ((unsigned)bf16_rne(hi) << 16);
}

// LDS byte offset of a __shared__ object (generic -> AS3 -> u32).
__device__ __forceinline__ unsigned lds_off(const void* p) {
  return (unsigned)(unsigned long long)(__attribute__((address_space(3))) const void*)p;
}

#if USE_ASYNC_COPY
// Async DMA: 16B per lane, global -> LDS, tracked by ASYNCcnt.
__device__ __forceinline__ void async_g2l_b128(unsigned lds_byte_off,
                                               unsigned gvoff_bytes,
                                               const void* sbase) {
  asm volatile("global_load_async_to_lds_b128 %0, %1, %2"
               :
               : "v"(lds_byte_off), "v"(gvoff_bytes), "s"(sbase)
               : "memory");
}
__device__ __forceinline__ void wait_async0() {
  asm volatile("s_wait_asynccnt 0x0" ::: "memory");
}
#endif

// ---------------------------------------------------------------------------
// K1: RMSNorm + router logits + top-4 + softmax.
// One block per token (256 threads = 8 wave32).
// ---------------------------------------------------------------------------
__global__ __launch_bounds__(256) void k_rms_router(
    const float* __restrict__ x, const float* __restrict__ rmsw,
    const float* __restrict__ rw, const float* __restrict__ rb,
    unsigned short* __restrict__ xn, int* __restrict__ topi,
    float* __restrict__ affw, float* __restrict__ aff_full) {
  __shared__ float xs[H_];
  __shared__ float red[256];
  __shared__ float lg[E_];
  const int t = blockIdx.x;
  const int tid = threadIdx.x;

  float p = 0.f;
  for (int h = tid; h < H_; h += 256) {
    float v = x[(size_t)t * H_ + h];
    xs[h] = v;
    p += v * v;
  }
  red[tid] = p;
  __syncthreads();
  for (int s = 128; s > 0; s >>= 1) {
    if (tid < s) red[tid] += red[tid + s];
    __syncthreads();
  }
  const float scale = rsqrtf(red[0] * (1.0f / H_) + 1e-5f);
  for (int h = tid; h < H_; h += 256) {
    float v = xs[h] * scale * rmsw[h];
    xs[h] = v;                                  // fp32 copy for router dot
    xn[(size_t)t * H_ + h] = bf16_rne(v);       // bf16 copy for WMMA path
  }
  __syncthreads();

  const int lane = tid & 31, wave = tid >> 5;
  for (int rep = 0; rep < 2; ++rep) {           // 8 waves x 2 = 16 experts
    const int e = wave * 2 + rep;
    float d = 0.f;
    for (int h = lane; h < H_; h += 32) d += xs[h] * rw[(size_t)e * H_ + h];
    for (int m = 16; m > 0; m >>= 1) d += __shfl_xor(d, m, 32);
    if (lane == 0) lg[e] = d + rb[e];
  }
  __syncthreads();

  if (tid == 0) {
    float lv[E_];
    for (int e = 0; e < E_; ++e) lv[e] = lg[e];
    int idx[K_]; float val[K_]; bool used[E_] = {};
    for (int k = 0; k < K_; ++k) {              // strict > keeps lowest index on ties
      float best = -3.4e38f; int bi = 0;
      for (int e = 0; e < E_; ++e)
        if (!used[e] && lv[e] > best) { best = lv[e]; bi = e; }
      used[bi] = true; idx[k] = bi; val[k] = best;
    }
    const float mx = val[0];
    float s = 0.f, a[K_];
    for (int k = 0; k < K_; ++k) { a[k] = __expf(val[k] - mx); s += a[k]; }
    const float inv = 1.f / s;
    for (int e = 0; e < E_; ++e) aff_full[t * E_ + e] = 0.f;
    for (int k = 0; k < K_; ++k) {
      a[k] *= inv;
      aff_full[t * E_ + idx[k]] = a[k];
      topi[t * K_ + k] = idx[k];
      affw[t * K_ + k] = a[k];
    }
  }
}

// ---------------------------------------------------------------------------
// K2: deterministic capacity scan (reference order: token-major, k within tok)
// ---------------------------------------------------------------------------
__global__ void k_scan(const int* __restrict__ topi,
                       int* __restrict__ slot_token,
                       int* __restrict__ cidx) {
  const int tid = threadIdx.x;
  for (int i = tid; i < E_ * C_; i += 256) slot_token[i] = -1;
  __syncthreads();
  if (tid < E_) {
    const int e = tid;
    int cnt = 0;
    for (int p = 0; p < T_ * K_; ++p) {
      if (topi[p] == e) {
        if (cnt < C_) {
          slot_token[e * C_ + cnt] = p >> 2;    // token index
          cidx[p] = e * C_ + cnt;
        } else {
          cidx[p] = -1;                          // dropped (over capacity)
        }
        ++cnt;
      }
    }
  }
}

// ---------------------------------------------------------------------------
// K3: dispatch bf16 token rows into dense (E*C, H) tiles; zero empty slots.
// ---------------------------------------------------------------------------
__global__ void k_dispatch(const unsigned short* __restrict__ xn,
                           const int* __restrict__ slot_token,
                           unsigned short* __restrict__ Xe) {
  const int slot = blockIdx.x;
  const int tsrc = slot_token[slot];
  unsigned int* dst = (unsigned int*)(Xe + (size_t)slot * H_);
  if (tsrc >= 0) {
    const unsigned int* src = (const unsigned int*)(xn + (size_t)tsrc * H_);
    for (int i = threadIdx.x; i < H_ / 2; i += 256) dst[i] = src[i];
  } else {
    for (int i = threadIdx.x; i < H_ / 2; i += 256) dst[i] = 0u;
  }
}

// ---------------------------------------------------------------------------
// K4: gate_up GEMM + fused activation.
// Block tile: 128(M) x 64(N-gate, paired with N+I up). 8 waves = 4(M) x 2(N),
// each wave owns 2x2 WMMA 16x16 tiles for gate AND up.
// A tile (bf16) DMA'd global->LDS via async b128; B tile fp32->bf16 packed
// pairwise and stored as dwords into transposed, padded LDS.
// ---------------------------------------------------------------------------
__global__ __launch_bounds__(256) void k_gate_up(
    const unsigned short* __restrict__ Xe,  // [E*C, H] bf16
    const float* __restrict__ gw,           // [E, H, 2I] fp32
    const float* __restrict__ gub,          // [E, 2I] fp32
    unsigned short* __restrict__ Act) {     // [E*C, I] bf16
  __shared__ alignas(32) unsigned short sA[128 * 48];  // [m][k], pitch 48
  __shared__ alignas(32) unsigned short sB[128 * 48];  // [n][k]; 0..63 gate, 64..127 up
  const int nt0 = blockIdx.x * 64;
  const int m0  = blockIdx.y * 128;
  const int e   = blockIdx.z;
  const int tid = threadIdx.x;
  const int lane = tid & 31, wave = tid >> 5;
  const int wm = wave & 3, wn = wave >> 2;

  v8f gacc[2][2] = {{{0.f}}};
  v8f uacc[2][2] = {{{0.f}}};

  const size_t xbase = (size_t)(e * C_ + m0) * H_;
  const size_t wbase = (size_t)e * H_ * (2 * I_);

  // A-tile chunks: 128 rows x 64B = 512 x 16B chunks, 2 per thread.
  const int ar0 = tid >> 2,          ap0 = (tid & 3) * 16;          // chunk tid
  const int ar1 = (tid + 256) >> 2,  ap1 = ((tid + 256) & 3) * 16;  // chunk tid+256
  const unsigned ldsA0 = lds_off(&sA[0]) + (unsigned)(ar0 * 96 + ap0);
  const unsigned ldsA1 = lds_off(&sA[0]) + (unsigned)(ar1 * 96 + ap1);
  const unsigned gvo0  = (unsigned)(ar0 * (H_ * 2) + ap0);
  const unsigned gvo1  = (unsigned)(ar1 * (H_ * 2) + ap1);

  // B-tile: thread covers k-pair (kk,kk+1) x 4 n values (packed b32 LDS stores)
  const int kk  = (tid & 15) * 2;
  const int ng4 = (tid >> 4) * 4;

  for (int k0 = 0; k0 < H_; k0 += 32) {
    const unsigned short* abase = Xe + xbase + k0;   // uniform -> SGPR pair
#if USE_ASYNC_COPY
    async_g2l_b128(ldsA0, gvo0, abase);
    async_g2l_b128(ldsA1, gvo1, abase);
#else
    {
      const uint4* a0 = (const uint4*)((const char*)abase + gvo0);
      const uint4* a1 = (const uint4*)((const char*)abase + gvo1);
      *(uint4*)((char*)&sA[0] + (ar0 * 96 + ap0)) = *a0;
      *(uint4*)((char*)&sA[0] + (ar1 * 96 + ap1)) = *a1;
    }
#endif
    // stage B: two k-rows, gate + up, convert & pack to bf16 pairs
    const float* gp0 = gw + wbase + (size_t)(k0 + kk) * (2 * I_) + nt0 + ng4;
    __builtin_prefetch(gp0 + 32 * (2 * I_), 0, 1);   // next k-step tile
    float4 ga = ((const float4*)gp0)[0];
    float4 gb = ((const float4*)(gp0 + 2 * I_))[0];
    float4 ua = ((const float4*)(gp0 + I_))[0];
    float4 ub = ((const float4*)(gp0 + I_ + 2 * I_))[0];
    const float gav[4] = {ga.x, ga.y, ga.z, ga.w};
    const float gbv[4] = {gb.x, gb.y, gb.z, gb.w};
    const float uav[4] = {ua.x, ua.y, ua.z, ua.w};
    const float ubv[4] = {ub.x, ub.y, ub.z, ub.w};
#pragma unroll
    for (int j = 0; j < 4; ++j) {
      const int n = ng4 + j;
      *(unsigned*)&sB[n * 48 + kk]        = pack_bf16(gav[j], gbv[j]);
      *(unsigned*)&sB[(64 + n) * 48 + kk] = pack_bf16(uav[j], ubv[j]);
    }
#if USE_ASYNC_COPY
    wait_async0();
#endif
    __syncthreads();

    const int koff = (lane >> 4) * 16;   // lanes 0-15: K 0..15, lanes 16-31: K 16..31
    const int r16  = lane & 15;
    v16bf a[2], fbg[2], fbu[2];
#pragma unroll
    for (int mt = 0; mt < 2; ++mt)
      a[mt] = *(const v16bf*)&sA[(wm * 32 + mt * 16 + r16) * 48 + koff];
#pragma unroll
    for (int nt = 0; nt < 2; ++nt) {
      fbg[nt] = *(const v16bf*)&sB[(wn * 32 + nt * 16 + r16) * 48 + koff];
      fbu[nt] = *(const v16bf*)&sB[(64 + wn * 32 + nt * 16 + r16) * 48 + koff];
    }
#pragma unroll
    for (int mt = 0; mt < 2; ++mt)
#pragma unroll
      for (int nt = 0; nt < 2; ++nt) {
        gacc[mt][nt] = __builtin_amdgcn_wmma_f32_16x16x32_bf16(
            false, a[mt], false, fbg[nt], (short)0, gacc[mt][nt], false, false);
        uacc[mt][nt] = __builtin_amdgcn_wmma_f32_16x16x32_bf16(
            false, a[mt], false, fbu[nt], (short)0, uacc[mt][nt], false, false);
      }
    __syncthreads();
  }

  // epilogue: bias + clipped swiglu, write bf16 Act
  const int colb = lane & 15, rowoff = (lane >> 4) * 8;
#pragma unroll
  for (int nt = 0; nt < 2; ++nt) {
    const int col = nt0 + wn * 32 + nt * 16 + colb;
    const float biasg = gub[(size_t)e * (2 * I_) + col];
    const float biasu = gub[(size_t)e * (2 * I_) + I_ + col];
#pragma unroll
    for (int mt = 0; mt < 2; ++mt) {
      v8f g = gacc[mt][nt], u = uacc[mt][nt];
#pragma unroll
      for (int r = 0; r < 8; ++r) {
        const int c = m0 + wm * 32 + mt * 16 + rowoff + r;
        float gg = fminf(g[r] + biasg, LIMIT_);
        float uu = fminf(fmaxf(u[r] + biasu, -LIMIT_), LIMIT_);
        float sig = 1.0f / (1.0f + __expf(-ALPHA_ * gg));
        float act = (uu + 1.0f) * (gg * sig);
        Act[(size_t)(e * C_ + c) * I_ + col] = bf16_rne(act);
      }
    }
  }
}

// ---------------------------------------------------------------------------
// K5: down GEMM (bf16 Act x bf16(dw) -> fp32 y + bias). Same tiling.
// ---------------------------------------------------------------------------
__global__ __launch_bounds__(256) void k_down(
    const unsigned short* __restrict__ Act,  // [E*C, I] bf16
    const float* __restrict__ dw,            // [E, I, H] fp32
    const float* __restrict__ db,            // [E, H] fp32
    float* __restrict__ Y) {                 // [E*C, H] fp32
  __shared__ alignas(32) unsigned short sA[128 * 48];
  __shared__ alignas(32) unsigned short sB[64 * 48];
  const int nt0 = blockIdx.x * 64;
  const int m0  = blockIdx.y * 128;
  const int e   = blockIdx.z;
  const int tid = threadIdx.x;
  const int lane = tid & 31, wave = tid >> 5;
  const int wm = wave & 3, wn = wave >> 2;

  v8f acc[2][2] = {{{0.f}}};
  const size_t abase_el = (size_t)(e * C_ + m0) * I_;
  const size_t wbase = (size_t)e * I_ * H_;

  const int ar0 = tid >> 2,          ap0 = (tid & 3) * 16;
  const int ar1 = (tid + 256) >> 2,  ap1 = ((tid + 256) & 3) * 16;
  const unsigned ldsA0 = lds_off(&sA[0]) + (unsigned)(ar0 * 96 + ap0);
  const unsigned ldsA1 = lds_off(&sA[0]) + (unsigned)(ar1 * 96 + ap1);
  const unsigned gvo0  = (unsigned)(ar0 * (I_ * 2) + ap0);
  const unsigned gvo1  = (unsigned)(ar1 * (I_ * 2) + ap1);

  const int kk  = (tid & 15) * 2;
  const int ng4 = (tid >> 4) * 4;

  for (int k0 = 0; k0 < I_; k0 += 32) {
    const unsigned short* abase = Act + abase_el + k0;
#if USE_ASYNC_COPY
    async_g2l_b128(ldsA0, gvo0, abase);
    async_g2l_b128(ldsA1, gvo1, abase);
#else
    {
      const uint4* a0 = (const uint4*)((const char*)abase + gvo0);
      const uint4* a1 = (const uint4*)((const char*)abase + gvo1);
      *(uint4*)((char*)&sA[0] + (ar0 * 96 + ap0)) = *a0;
      *(uint4*)((char*)&sA[0] + (ar1 * 96 + ap1)) = *a1;
    }
#endif
    const float* bp0 = dw + wbase + (size_t)(k0 + kk) * H_ + nt0 + ng4;
    __builtin_prefetch(bp0 + 32 * H_, 0, 1);
    float4 b0 = ((const float4*)bp0)[0];
    float4 b1 = ((const float4*)(bp0 + H_))[0];
    const float b0v[4] = {b0.x, b0.y, b0.z, b0.w};
    const float b1v[4] = {b1.x, b1.y, b1.z, b1.w};
#pragma unroll
    for (int j = 0; j < 4; ++j)
      *(unsigned*)&sB[(ng4 + j) * 48 + kk] = pack_bf16(b0v[j], b1v[j]);
#if USE_ASYNC_COPY
    wait_async0();
#endif
    __syncthreads();

    const int koff = (lane >> 4) * 16;
    const int r16  = lane & 15;
    v16bf a[2], fb[2];
#pragma unroll
    for (int mt = 0; mt < 2; ++mt)
      a[mt] = *(const v16bf*)&sA[(wm * 32 + mt * 16 + r16) * 48 + koff];
#pragma unroll
    for (int nt = 0; nt < 2; ++nt)
      fb[nt] = *(const v16bf*)&sB[(wn * 32 + nt * 16 + r16) * 48 + koff];
#pragma unroll
    for (int mt = 0; mt < 2; ++mt)
#pragma unroll
      for (int nt = 0; nt < 2; ++nt)
        acc[mt][nt] = __builtin_amdgcn_wmma_f32_16x16x32_bf16(
            false, a[mt], false, fb[nt], (short)0, acc[mt][nt], false, false);
    __syncthreads();
  }

  const int colb = lane & 15, rowoff = (lane >> 4) * 8;
#pragma unroll
  for (int nt = 0; nt < 2; ++nt) {
    const int col = nt0 + wn * 32 + nt * 16 + colb;
    const float bias = db[(size_t)e * H_ + col];
#pragma unroll
    for (int mt = 0; mt < 2; ++mt) {
      v8f v = acc[mt][nt];
#pragma unroll
      for (int r = 0; r < 8; ++r) {
        const int c = m0 + wm * 32 + mt * 16 + rowoff + r;
        Y[(size_t)(e * C_ + c) * H_ + col] = v[r] + bias;
      }
    }
  }
}

// ---------------------------------------------------------------------------
// K6: weighted combine of up-to-4 expert outputs per token.
// ---------------------------------------------------------------------------
__global__ void k_combine(const float* __restrict__ Y,
                          const int* __restrict__ cidx,
                          const float* __restrict__ affw,
                          float* __restrict__ out) {
  const int t = blockIdx.x;
  int id[K_]; float aw[K_];
#pragma unroll
  for (int k = 0; k < K_; ++k) { id[k] = cidx[t * K_ + k]; aw[k] = affw[t * K_ + k]; }
  for (int h = threadIdx.x; h < H_; h += 256) {
    float acc = 0.f;
#pragma unroll
    for (int k = 0; k < K_; ++k)
      if (id[k] >= 0) acc += aw[k] * Y[(size_t)id[k] * H_ + h];
    out[(size_t)t * H_ + h] = acc;
  }
}

// ---------------------------------------------------------------------------
extern "C" void kernel_launch(void* const* d_in, const int* in_sizes, int n_in,
                              void* d_out, int out_size, void* d_ws, size_t ws_size,
                              hipStream_t stream) {
  (void)in_sizes; (void)n_in; (void)out_size; (void)ws_size;
  const float* x    = (const float*)d_in[0];
  const float* rmsw = (const float*)d_in[1];
  const float* rw   = (const float*)d_in[2];
  const float* rb   = (const float*)d_in[3];
  const float* gw   = (const float*)d_in[4];
  const float* gub  = (const float*)d_in[5];
  const float* dw   = (const float*)d_in[6];
  const float* db   = (const float*)d_in[7];

  float* out_comb = (float*)d_out;                      // [T, H]
  float* aff_full = out_comb + (size_t)T_ * H_;         // [T, E]

  // bump-allocate workspace (~104 MB total)
  char* ws = (char*)d_ws;
  size_t off = 0;
  auto alloc = [&](size_t bytes) -> void* {
    void* p = ws + off;
    off = (off + bytes + 255) & ~(size_t)255;
    return p;
  };
  unsigned short* xn  = (unsigned short*)alloc((size_t)T_ * H_ * 2);       // 3 MB
  int*   topi         = (int*)  alloc((size_t)T_ * K_ * 4);
  float* affw         = (float*)alloc((size_t)T_ * K_ * 4);
  int*   slot_token   = (int*)  alloc((size_t)E_ * C_ * 4);
  int*   cidx         = (int*)  alloc((size_t)T_ * K_ * 4);
  unsigned short* Xe  = (unsigned short*)alloc((size_t)E_ * C_ * H_ * 2);  // 25 MB
  unsigned short* Act = (unsigned short*)alloc((size_t)E_ * C_ * I_ * 2);  // 25 MB
  float* Y            = (float*)alloc((size_t)E_ * C_ * H_ * 4);           // 50 MB

  k_rms_router<<<T_, 256, 0, stream>>>(x, rmsw, rw, rb, xn, topi, affw, aff_full);
  k_scan<<<1, 256, 0, stream>>>(topi, slot_token, cidx);
  k_dispatch<<<E_ * C_, 256, 0, stream>>>(xn, slot_token, Xe);
  dim3 g1(I_ / 64, C_ / 128, E_);
  k_gate_up<<<g1, 256, 0, stream>>>(Xe, gw, gub, Act);
  dim3 g2(H_ / 64, C_ / 128, E_);
  k_down<<<g2, 256, 0, stream>>>(Act, dw, db, Y);
  k_combine<<<T_, 256, 0, stream>>>(Y, cidx, affw, out_comb);
}
